// RTreeNet_20014547599562
// MI455X (gfx1250) — matrix operand order repaired and checked by reading, not compiled, not run on hardware
//
#include <hip/hip_runtime.h>
#include <hip/hip_bf16.h>
#include <math.h>

#define BB 8
#define NN 8192
#define GG 1024
#define KK 32
#define HEADS 4

typedef __attribute__((ext_vector_type(16))) _Float16 v16h;
typedef __attribute__((ext_vector_type(8)))  float    v8f;

union FragU { v16h h; uint4 u[2]; };

__device__ __forceinline__ v16h load_frag_pair(const _Float16* p0, const _Float16* p1) {
  FragU f;
  f.u[0] = *reinterpret_cast<const uint4*>(p0);
  f.u[1] = *reinterpret_cast<const uint4*>(p1);
  return f.h;
}

__device__ __forceinline__ v8f wmma_f16(v16h a, v16h b, v8f c) {
  return __builtin_amdgcn_wmma_f32_16x16x32_f16(false, a, false, b, (short)0, c, false, false);
}

// ---------------- FPS: one block per batch, 32 pts/thread in registers ----------------
__global__ void __launch_bounds__(256) fps_kernel(const float* __restrict__ xyz,
                                                  int* __restrict__ fps_idx) {
  const int b = blockIdx.x;
  const int tid = threadIdx.x;
  const float* base = xyz + (size_t)b * NN * 3;
  float px[32], py[32], pz[32], dd[32];
#pragma unroll
  for (int i = 0; i < 32; ++i) {
    int p = tid + 256 * i;
    px[i] = base[p * 3 + 0]; py[i] = base[p * 3 + 1]; pz[i] = base[p * 3 + 2];
    dd[i] = 1e30f;
  }
  __shared__ float s_val[8];
  __shared__ int   s_idx[8];
  __shared__ int   s_win;
  if (tid == 0) fps_idx[b * GG] = 0;
  int cur = 0;
  const int lane = tid & 31, wid = tid >> 5;
  for (int it = 1; it < GG; ++it) {
    float cx = base[cur * 3 + 0], cy = base[cur * 3 + 1], cz = base[cur * 3 + 2];
    float bv = -1.0f; int bi = 0x7fffffff;
#pragma unroll
    for (int i = 0; i < 32; ++i) {
      float dx = px[i] - cx, dy = py[i] - cy, dz = pz[i] - cz;
      float d = dx * dx + dy * dy + dz * dz;
      d = fminf(dd[i], d);
      dd[i] = d;
      if (d > bv) { bv = d; bi = tid + 256 * i; }
    }
#pragma unroll
    for (int off = 16; off > 0; off >>= 1) {
      float ov = __shfl_xor(bv, off, 32);
      int   oi = __shfl_xor(bi, off, 32);
      if (ov > bv || (ov == bv && oi < bi)) { bv = ov; bi = oi; }
    }
    if (lane == 0) { s_val[wid] = bv; s_idx[wid] = bi; }
    __syncthreads();
    if (tid == 0) {
      float v = s_val[0]; int ix = s_idx[0];
      for (int w = 1; w < 8; ++w)
        if (s_val[w] > v || (s_val[w] == v && s_idx[w] < ix)) { v = s_val[w]; ix = s_idx[w]; }
      s_win = ix;
      fps_idx[b * GG + it] = ix;
    }
    __syncthreads();
    cur = s_win;
  }
}

// ---------------- gather lc_xyz / lc_x ----------------
__global__ void gather_kernel(const float* __restrict__ xyz, const float* __restrict__ x,
                              const int* __restrict__ fps_idx,
                              float* __restrict__ lc_xyz, float* __restrict__ lc_x) {
  int gid = blockIdx.x * blockDim.x + threadIdx.x;
  if (gid < BB * GG * 3) {
    int bg = gid / 3, c = gid - bg * 3;
    int b = bg >> 10;
    int idx = fps_idx[bg];
    lc_xyz[gid] = xyz[((size_t)b * NN + idx) * 3 + c];
  }
  if (gid < BB * GG * 64) {
    int bg = gid >> 6, c = gid & 63;
    int b = bg >> 10;
    int idx = fps_idx[bg];
    lc_x[gid] = x[((size_t)b * NN + idx) * 64 + c];
  }
}

// ---------------- KNN top-32 (iterative min with tie->lowest index) ----------------
__global__ void __launch_bounds__(256) knn_kernel(const float* __restrict__ xyz,
                                                  const float* __restrict__ lc_xyz,
                                                  int* __restrict__ knn_idx) {
  const int bg = blockIdx.x;
  const int b = bg >> 10;
  const int tid = threadIdx.x;
  const float qx = lc_xyz[bg * 3 + 0], qy = lc_xyz[bg * 3 + 1], qz = lc_xyz[bg * 3 + 2];
  const float* base = xyz + (size_t)b * NN * 3;
  float dd[32];
#pragma unroll
  for (int i = 0; i < 32; ++i) {
    int p = tid + 256 * i;
    float dx = base[p * 3 + 0] - qx, dy = base[p * 3 + 1] - qy, dz = base[p * 3 + 2] - qz;
    dd[i] = dx * dx + dy * dy + dz * dz;
  }
  __shared__ float s_val[8];
  __shared__ int   s_idx[8];
  __shared__ int   s_win;
  const int lane = tid & 31, wid = tid >> 5;
  for (int kk = 0; kk < KK; ++kk) {
    float bv = 1e38f; int bi = 0x7fffffff;
#pragma unroll
    for (int i = 0; i < 32; ++i)
      if (dd[i] < bv) { bv = dd[i]; bi = tid + 256 * i; }
#pragma unroll
    for (int off = 16; off > 0; off >>= 1) {
      float ov = __shfl_xor(bv, off, 32);
      int   oi = __shfl_xor(bi, off, 32);
      if (ov < bv || (ov == bv && oi < bi)) { bv = ov; bi = oi; }
    }
    if (lane == 0) { s_val[wid] = bv; s_idx[wid] = bi; }
    __syncthreads();
    if (tid == 0) {
      float v = s_val[0]; int ix = s_idx[0];
      for (int w = 1; w < 8; ++w)
        if (s_val[w] < v || (s_val[w] == v && s_idx[w] < ix)) { v = s_val[w]; ix = s_idx[w]; }
      s_win = ix;
      knn_idx[bg * KK + kk] = ix;
    }
    __syncthreads();
    int wI = s_win;
    if ((wI & 255) == tid) dd[wI >> 8] = 1e38f;
  }
}

// ---------------- global scalar std reductions ----------------
__global__ void init_stats(float* stats) { if (threadIdx.x < 8) stats[threadIdx.x] = 0.f; }

__global__ void __launch_bounds__(256) stats_x_kernel(const float* __restrict__ x,
                                                      const float* __restrict__ lc_x,
                                                      const int* __restrict__ knn_idx,
                                                      float* __restrict__ stats) {
  const long total = (long)BB * GG * KK * 64;
  long gid = (long)blockIdx.x * 256 + threadIdx.x;
  long stride = (long)gridDim.x * 256;
  float s = 0.f, sq = 0.f;
  for (long i = gid; i < total; i += stride) {
    int c = (int)(i & 63);
    int k = (int)((i >> 6) & 31);
    int bg = (int)(i >> 11);
    int b = bg >> 10;
    int n = knn_idx[bg * KK + k];
    float v = x[((size_t)b * NN + n) * 64 + c] - lc_x[bg * 64 + c];
    s += v; sq += v * v;
  }
  __shared__ float rs[256], rq[256];
  int tid = threadIdx.x;
  rs[tid] = s; rq[tid] = sq;
  __syncthreads();
  for (int off = 128; off > 0; off >>= 1) {
    if (tid < off) { rs[tid] += rs[tid + off]; rq[tid] += rq[tid + off]; }
    __syncthreads();
  }
  if (tid == 0) { atomicAdd(&stats[0], rs[0]); atomicAdd(&stats[1], rq[0]); }
}

__global__ void __launch_bounds__(256) stats_xyz_kernel(const float* __restrict__ xyz,
                                                        const float* __restrict__ lc_xyz,
                                                        const int* __restrict__ knn_idx,
                                                        float* __restrict__ stats) {
  const int total = BB * GG * KK * 3;
  int gid = blockIdx.x * 256 + threadIdx.x;
  int stride = gridDim.x * 256;
  float s = 0.f, sq = 0.f;
  for (int i = gid; i < total; i += stride) {
    int c = i % 3;
    int k = (i / 3) & 31;
    int bg = i / 96;
    int b = bg >> 10;
    int n = knn_idx[bg * KK + k];
    float v = xyz[((size_t)b * NN + n) * 3 + c] - lc_xyz[bg * 3 + c];
    s += v; sq += v * v;
  }
  __shared__ float rs[256], rq[256];
  int tid = threadIdx.x;
  rs[tid] = s; rq[tid] = sq;
  __syncthreads();
  for (int off = 128; off > 0; off >>= 1) {
    if (tid < off) { rs[tid] += rs[tid + off]; rq[tid] += rq[tid + off]; }
    __syncthreads();
  }
  if (tid == 0) { atomicAdd(&stats[2], rs[0]); atomicAdd(&stats[3], rq[0]); }
}

__global__ void finalize_stats(float* stats) {
  if (threadIdx.x == 0) {
    double nx = (double)BB * GG * KK * 64;
    double var = ((double)stats[1] - (double)stats[0] * (double)stats[0] / nx) / (nx - 1.0);
    stats[4] = (float)(1.0 / (sqrt(fmax(var, 0.0)) + 1e-5));
    double nz = (double)BB * GG * KK * 3;
    double vz = ((double)stats[3] - (double)stats[2] * (double)stats[2] / nz) / (nz - 1.0);
    stats[5] = (float)(1.0 / (sqrt(fmax(vz, 0.0)) + 1e-5));
  }
}

// ---------------- LGA + positional embedding + max over k -> f16 (b,n,128) ----------------
__global__ void __launch_bounds__(128) lga_kernel(const float* __restrict__ xyz,
                                                  const float* __restrict__ x,
                                                  const float* __restrict__ lc_xyz,
                                                  const float* __restrict__ lc_x,
                                                  const int* __restrict__ knn_idx,
                                                  const float* __restrict__ stats,
                                                  _Float16* __restrict__ xh) {
  const int bg = blockIdx.x;
  const int b = bg >> 10;
  const int c = threadIdx.x;
  const float invsx = stats[4], invsz = stats[5];
  const float qx = lc_xyz[bg * 3 + 0], qy = lc_xyz[bg * 3 + 1], qz = lc_xyz[bg * 3 + 2];
  int d = c / 42;
  int r = c - d * 42;
  int f = r >> 1;
  int sflag = r & 1;
  float dimdiv = 100.0f * powf(1000.0f, -(float)f / 21.0f);  // BETA * ALPHA^(-f/feat_dim)
  float lcc = (c < 64) ? lc_x[bg * 64 + c] : lc_x[bg * 64 + (c & 63)];
  float m = -1e30f;
  const float* base = xyz + (size_t)b * NN * 3;
  for (int k = 0; k < KK; ++k) {
    int nidx = knn_idx[bg * KK + k];
    float nx = (base[nidx * 3 + 0] - qx) * invsz;
    float ny = (base[nidx * 3 + 1] - qy) * invsz;
    float nz = (base[nidx * 3 + 2] - qz) * invsz;
    float coord = (d == 0) ? nx : ((d == 1) ? ny : nz);
    float dv = coord * dimdiv;
    float pe = sflag ? __cosf(dv) : __sinf(dv);
    float kx = (c < 64) ? (x[((size_t)b * NN + nidx) * 64 + c] - lcc) * invsx : lcc;
    float v = (c < 126) ? (kx + pe) * pe : 0.0f;
    m = fmaxf(m, v);
  }
  xh[(size_t)bg * 128 + c] = (_Float16)m;
}

// ---------------- weight conversion ----------------
__global__ void convw_kernel(const float* __restrict__ wq, const float* __restrict__ wp,
                             _Float16* __restrict__ wqh, _Float16* __restrict__ wph) {
  int i = blockIdx.x * 256 + threadIdx.x;
  if (i < 384 * 128) wqh[i] = (_Float16)wq[i];
  if (i < 128 * 128) wph[i] = (_Float16)wp[i];
}

// ---------------- QKV GEMM: (8192x128) x (384x128)^T via WMMA ----------------
__global__ void __launch_bounds__(128) qkv_gemm_kernel(const _Float16* __restrict__ xh,
                                                       const _Float16* __restrict__ wqh,
                                                       _Float16* __restrict__ qkv) {
  const int wid = blockIdx.x * 4 + (threadIdx.x >> 5);
  const int lane = threadIdx.x & 31;
  const int NT = 24;
  const int mtile = wid / NT;
  const int ntile = wid - mtile * NT;
  const int ml = lane & 15;
  const int hi = lane >> 4;
  const _Float16* arow = xh + (size_t)(mtile * 16 + ml) * 128;
  const _Float16* brow = wqh + (size_t)(ntile * 16 + ml) * 128;
  v8f acc = {};
#pragma unroll
  for (int ks = 0; ks < 4; ++ks) {
    v16h a = load_frag_pair(arow + ks * 32 + hi * 8, arow + ks * 32 + hi * 8 + 16);
    v16h w = load_frag_pair(brow + ks * 32 + hi * 8, brow + ks * 32 + hi * 8 + 16);
    acc = wmma_f16(a, w, acc);
  }
#pragma unroll
  for (int r = 0; r < 8; ++r) {
    int row = mtile * 16 + r + hi * 8;
    int col = ntile * 16 + ml;
    qkv[(size_t)row * 384 + col] = (_Float16)acc[r];
  }
}

// ---------------- V transpose: vt[b,h,dh,n] ----------------
__global__ void vt_kernel(const _Float16* __restrict__ qkv, _Float16* __restrict__ vt) {
  int i = blockIdx.x * 256 + threadIdx.x;
  int n = i & 1023, dh = (i >> 10) & 31, h = (i >> 15) & 3, b = i >> 17;
  vt[i] = qkv[((size_t)(b * GG + n)) * 384 + 256 + h * 32 + dh];
}

// ---------------- attention: 1 wave per (b,h,16-row tile), LDS score buffer ----------------
__global__ void __launch_bounds__(32) attn_kernel(const _Float16* __restrict__ qkv,
                                                  const _Float16* __restrict__ vt,
                                                  _Float16* __restrict__ ao) {
  __shared__ __align__(16) _Float16 Sh[16 * 1024];
  __shared__ float rsum[16];
  const int blk = blockIdx.x;
  const int mt = blk & 63;
  const int h = (blk >> 6) & 3;
  const int b = blk >> 8;
  const int lane = threadIdx.x;
  const int ml = lane & 15;
  const int hi = lane >> 4;
  const float scale = 0.17677669529663687f;  // 32^-0.5
  const _Float16* qrow = qkv + ((size_t)(b * GG + mt * 16 + ml)) * 384 + h * 32;
  v16h aq = load_frag_pair(qrow + hi * 8, qrow + hi * 8 + 16);
  for (int jt = 0; jt < 64; ++jt) {
    const _Float16* krow = qkv + ((size_t)(b * GG + jt * 16 + ml)) * 384 + 128 + h * 32;
    v16h bk = load_frag_pair(krow + hi * 8, krow + hi * 8 + 16);
    v8f z = {};
    v8f s = wmma_f16(aq, bk, z);
#pragma unroll
    for (int r = 0; r < 8; ++r)
      Sh[(r + hi * 8) * 1024 + jt * 16 + ml] = (_Float16)(s[r] * scale);
  }
  __syncthreads();
  {
    int row = lane >> 1;
    int half = lane & 1;
    _Float16* rp = Sh + row * 1024 + half * 512;
    float mx = -1e30f;
    for (int i = 0; i < 512; ++i) mx = fmaxf(mx, (float)rp[i]);
    mx = fmaxf(mx, __shfl_xor(mx, 1, 32));
    float sum = 0.f;
    for (int i = 0; i < 512; ++i) {
      float e = __expf((float)rp[i] - mx);
      sum += e;
      rp[i] = (_Float16)e;
    }
    sum += __shfl_xor(sum, 1, 32);
    if (half == 0) rsum[row] = sum;
  }
  __syncthreads();
  v8f acc0 = {}, acc1 = {};
  const _Float16* vbase = vt + (size_t)(b * HEADS + h) * 32 * 1024;
  for (int jt2 = 0; jt2 < 32; ++jt2) {
    const _Float16* prow = Sh + ml * 1024 + jt2 * 32 + hi * 8;
    v16h ap = load_frag_pair(prow, prow + 16);
    const _Float16* v0 = vbase + (size_t)ml * 1024 + jt2 * 32 + hi * 8;
    const _Float16* v1 = vbase + (size_t)(16 + ml) * 1024 + jt2 * 32 + hi * 8;
    acc0 = wmma_f16(ap, load_frag_pair(v0, v0 + 16), acc0);
    acc1 = wmma_f16(ap, load_frag_pair(v1, v1 + 16), acc1);
  }
#pragma unroll
  for (int r = 0; r < 8; ++r) {
    int m = r + hi * 8;
    float inv = 1.0f / rsum[m];
    size_t row = (size_t)(b * GG + mt * 16 + m) * 128 + h * 32;
    ao[row + ml] = (_Float16)(acc0[r] * inv);
    ao[row + 16 + ml] = (_Float16)(acc1[r] * inv);
  }
}

// ---------------- projection GEMM + bias -> f32 ----------------
__global__ void __launch_bounds__(128) proj_gemm_kernel(const _Float16* __restrict__ ao,
                                                        const _Float16* __restrict__ wph,
                                                        const float* __restrict__ bias,
                                                        float* __restrict__ t) {
  const int wid = blockIdx.x * 4 + (threadIdx.x >> 5);
  const int lane = threadIdx.x & 31;
  const int NT = 8;
  const int mtile = wid / NT;
  const int ntile = wid - mtile * NT;
  const int ml = lane & 15;
  const int hi = lane >> 4;
  const _Float16* arow = ao + (size_t)(mtile * 16 + ml) * 128;
  const _Float16* brow = wph + (size_t)(ntile * 16 + ml) * 128;
  v8f acc = {};
#pragma unroll
  for (int ks = 0; ks < 4; ++ks) {
    v16h a = load_frag_pair(arow + ks * 32 + hi * 8, arow + ks * 32 + hi * 8 + 16);
    v16h w = load_frag_pair(brow + ks * 32 + hi * 8, brow + ks * 32 + hi * 8 + 16);
    acc = wmma_f16(a, w, acc);
  }
#pragma unroll
  for (int r = 0; r < 8; ++r) {
    int row = mtile * 16 + r + hi * 8;
    int col = ntile * 16 + ml;
    t[(size_t)row * 128 + col] = acc[r] + bias[col];
  }
}

// ---------------- CCTA (collapsed): out[b,c,n] = exp(softmax_n(t)) + 1 + t ----------------
__global__ void __launch_bounds__(256) ccta_kernel(const float* __restrict__ t,
                                                   float* __restrict__ out) {
  __shared__ float red[256];
  const int b = blockIdx.x >> 7;
  const int c = blockIdx.x & 127;
  const int tid = threadIdx.x;
  const float* tb = t + (size_t)b * GG * 128 + c;
  float v[4];
#pragma unroll
  for (int j = 0; j < 4; ++j) v[j] = tb[(size_t)(tid + j * 256) * 128];
  float mx = fmaxf(fmaxf(v[0], v[1]), fmaxf(v[2], v[3]));
  red[tid] = mx; __syncthreads();
  for (int off = 128; off > 0; off >>= 1) {
    if (tid < off) red[tid] = fmaxf(red[tid], red[tid + off]);
    __syncthreads();
  }
  mx = red[0]; __syncthreads();
  float e[4]; float sum = 0.f;
#pragma unroll
  for (int j = 0; j < 4; ++j) { e[j] = __expf(v[j] - mx); sum += e[j]; }
  red[tid] = sum; __syncthreads();
  for (int off = 128; off > 0; off >>= 1) {
    if (tid < off) red[tid] += red[tid + off];
    __syncthreads();
  }
  float inv = 1.0f / red[0];
  size_t obase = ((size_t)b * 128 + c) * GG;
#pragma unroll
  for (int j = 0; j < 4; ++j) {
    int n = tid + j * 256;
    out[obase + n] = __expf(e[j] * inv) + 1.0f + v[j];
  }
}

extern "C" void kernel_launch(void* const* d_in, const int* in_sizes, int n_in,
                              void* d_out, int out_size, void* d_ws, size_t ws_size,
                              hipStream_t stream) {
  const float* xyz = (const float*)d_in[0];
  const float* x   = (const float*)d_in[1];
  const float* wq  = (const float*)d_in[2];
  const float* wp  = (const float*)d_in[3];
  const float* pb  = (const float*)d_in[4];
  char* ws = (char*)d_ws;
  int*      fps_idx = (int*)(ws + 0);              //  32 KB
  float*    lc_xyz  = (float*)(ws + 32768);        //  96 KB
  float*    lc_x    = (float*)(ws + 131072);       //   2 MB
  int*      knn_idx = (int*)(ws + 2228224);        //   1 MB
  float*    stats   = (float*)(ws + 3276800);      //  256 B
  _Float16* xh      = (_Float16*)(ws + 3277056);   //   2 MB
  _Float16* wqh     = (_Float16*)(ws + 5374208);   //  96 KB
  _Float16* wph     = (_Float16*)(ws + 5472512);   //  32 KB
  _Float16* qkv     = (_Float16*)(ws + 5505280);   //   6 MB
  _Float16* vt      = (_Float16*)(ws + 11796736);  //   2 MB
  _Float16* ao      = (_Float16*)(ws + 13893888);  //   2 MB
  float*    tbuf    = (float*)(ws + 15991040);     //   4 MB
  float* outp = (float*)d_out;

  init_stats<<<1, 32, 0, stream>>>(stats);
  fps_kernel<<<BB, 256, 0, stream>>>(xyz, fps_idx);
  gather_kernel<<<2048, 256, 0, stream>>>(xyz, x, fps_idx, lc_xyz, lc_x);
  knn_kernel<<<BB * GG, 256, 0, stream>>>(xyz, lc_xyz, knn_idx);
  stats_x_kernel<<<4096, 256, 0, stream>>>(x, lc_x, knn_idx, stats);
  stats_xyz_kernel<<<768, 256, 0, stream>>>(xyz, lc_xyz, knn_idx, stats);
  finalize_stats<<<1, 32, 0, stream>>>(stats);
  lga_kernel<<<BB * GG, 128, 0, stream>>>(xyz, x, lc_xyz, lc_x, knn_idx, stats, xh);
  convw_kernel<<<192, 256, 0, stream>>>(wq, wp, wqh, wph);
  qkv_gemm_kernel<<<3072, 128, 0, stream>>>(xh, wqh, qkv);
  vt_kernel<<<4096, 256, 0, stream>>>(qkv, vt);
  attn_kernel<<<BB * HEADS * 64, 32, 0, stream>>>(qkv, vt, ao);
  proj_gemm_kernel<<<1024, 128, 0, stream>>>(ao, wph, pb, tbuf);
  ccta_kernel<<<BB * 128, 256, 0, stream>>>(tbuf, outp);
}
